// SelfAttention_3298534883432
// MI455X (gfx1250) — matrix operand order repaired and checked by reading, not compiled, not run on hardware
//
#include <hip/hip_runtime.h>
#include <hip/hip_bf16.h>

// ---------------------------------------------------------------------------
// Causal multi-head self-attention, MI455X (gfx1250), wave32 + WMMA f16/f32.
// v3: f16-preconverted weights + double-buffered async global->LDS staging
// (GLOBAL_LOAD_ASYNC_TO_LDS_B128 / ASYNCcnt) in the GEMM kernels.
// ---------------------------------------------------------------------------

typedef _Float16 half_t;
typedef __attribute__((ext_vector_type(16))) _Float16 v16h;
typedef __attribute__((ext_vector_type(8)))  float    v8f;

#define WMMA_F16(a, b, c) \
    __builtin_amdgcn_wmma_f32_16x16x32_f16(false, (a), false, (b), (short)0, (c), false, false)

__device__ __forceinline__ v8f zero8() {
    v8f z;
#pragma unroll
    for (int i = 0; i < 8; ++i) z[i] = 0.0f;
    return z;
}

// Same-wave DS ordering fence (DS ops are in-order per ISA 7.3).
__device__ __forceinline__ void lds_fence_wave() {
    asm volatile("s_wait_dscnt 0x0" ::: "memory");
}

// Async global->LDS copy, 16B per lane (tracked with ASYNCcnt).
__device__ __forceinline__ void async_copy_b128(unsigned lds_off, const half_t* gaddr) {
    asm volatile("global_load_async_to_lds_b128 %0, %1, off"
                 :: "v"(lds_off), "v"(gaddr) : "memory");
}
__device__ __forceinline__ void wait_async_le3() {
    asm volatile("s_wait_asynccnt 0x3" ::: "memory");
}
__device__ __forceinline__ void wait_async_le1() {
    asm volatile("s_wait_asynccnt 0x1" ::: "memory");
}
__device__ __forceinline__ void wait_async_0() {
    asm volatile("s_wait_asynccnt 0x0" ::: "memory");
}

// LDS byte offset of a __shared__ object (LDS aperture: low 32 bits of the address).
__device__ __forceinline__ unsigned lds_offset(const void* p) {
    return (unsigned)(uintptr_t)p;
}

// A fragment: 16x32 (MxK) f16, source row-major, leading dim ld (halves).
__device__ __forceinline__ v16h load_a_f16(const half_t* base, int ld) {
    const int lane = threadIdx.x & 31;
    const int row  = lane & 15;
    const int kb   = (lane >> 4) << 3;   // 0 or 8
    const half_t* p = base + row * ld;
    v16h a;
#pragma unroll
    for (int e = 0; e < 8; ++e) a[e]     = p[kb + e];
#pragma unroll
    for (int e = 0; e < 8; ++e) a[8 + e] = p[16 + kb + e];
    return a;
}

// B fragment: 32x16 (KxN) f16, source stored as [n][k] row-major (B^T).
__device__ __forceinline__ v16h load_b_f16(const half_t* base, int ld) {
    const int lane = threadIdx.x & 31;
    const half_t* p = base + (lane & 15) * ld + ((lane >> 4) << 3);
    v16h b;
#pragma unroll
    for (int e = 0; e < 8; ++e) b[e] = p[e];
#pragma unroll
    for (int e = 0; e < 8; ++e) b[8 + e] = p[16 + e];
    return b;
}

static constexpr int Bc = 2, Tc = 2048, Cc = 1024, Hc = 16, Dh = 64;

// ---------------------------------------------------------------------------
// Pre-pass kernels: one-shot f32->f16 conversion of x and Wo; Wq/Wk/Wv
// additionally transposed per head to [d][c] so B-fragments are contiguous.
// ---------------------------------------------------------------------------
__global__ __launch_bounds__(256) void cvt_f16_kernel(
    const float* __restrict__ in, half_t* __restrict__ out, int n) {
    const int i = blockIdx.x * 256 + threadIdx.x;
    if (i < n) out[i] = (half_t)in[i];
}

// W: [16][1024][64] f32  ->  out: [16][64][1024] f16
__global__ __launch_bounds__(256) void transpose_w_kernel(
    const float* __restrict__ W, half_t* __restrict__ out) {
    const int idx = blockIdx.x * 256 + threadIdx.x;   // over 16*64*1024
    const int h   = idx >> 16;
    const int r   = idx & 65535;
    const int d   = r >> 10;
    const int c   = r & 1023;
    out[idx] = (half_t)W[((size_t)h * Cc + c) * Dh + d];
}

// ---------------------------------------------------------------------------
// Kernel 1: QKV projection. grid = (B*T/128, H), block = 256 (8 waves).
// wall: f16 [mat][H][64][1024] (pre-transposed). Per k-step a 192x32 tile
// (12KB) is async-copied to LDS, double-buffered; 8 waves x 3 asyncs each.
// q,k: f16 [BH][T][64]; vt: f16 [BH][64][T]. Scale 1/32 folded into Q.
// ---------------------------------------------------------------------------
__global__ __launch_bounds__(256) void qkv_kernel(
    const half_t* __restrict__ xh, const half_t* __restrict__ wall,
    half_t* __restrict__ q, half_t* __restrict__ k, half_t* __restrict__ vt) {

    const int h      = blockIdx.y;
    const int wave   = threadIdx.x >> 5;
    const int m_glob = blockIdx.x * 128 + wave * 16;     // flat row into [B*T]
    const int b      = m_glob >> 11;
    const int t0     = m_glob & 2047;
    const int bh     = b * Hc + h;

    constexpr int TILE = 192 * 32;                        // halves per buffer
    __shared__ half_t Wlds[2 * TILE];                     // 24KB double buffer

    v8f acc[12];
#pragma unroll
    for (int i = 0; i < 12; ++i) acc[i] = zero8();

    const unsigned lds0 = lds_offset(Wlds);

    // issue async stage of a 192x32 tile into buffer `buf` for k-offset c0
    auto stage = [&](int buf, int c0) {
#pragma unroll
        for (int i = 0; i < 3; ++i) {
            const int f   = i * 256 + threadIdx.x;        // 0..767 chunk id
            const int row = f >> 2;                        // 0..191
            const int seg = f & 3;
            const int m   = row >> 6;                      // mat 0..2
            const int d   = row & 63;
            const half_t* g = wall + ((size_t)(m * 1024 + h * 64 + d)) * Cc
                                   + c0 + seg * 8;
            async_copy_b128(lds0 + (unsigned)(buf * TILE + row * 32 + seg * 8) * 2, g);
        }
    };

    stage(0, 0);
    for (int s = 0; s < 32; ++s) {
        if (s + 1 < 32) { stage((s + 1) & 1, (s + 1) * 32); wait_async_le3(); }
        else            { wait_async_0(); }
        __syncthreads();

        const half_t* Wcur = Wlds + (s & 1) * TILE;
        const v16h a = load_a_f16(xh + (size_t)m_glob * Cc + s * 32, Cc);
#pragma unroll
        for (int m = 0; m < 3; ++m)
#pragma unroll
            for (int dt = 0; dt < 4; ++dt) {
                const v16h bf = load_b_f16(Wcur + (m * 64 + dt * 16) * 32, 32);
                acc[m * 4 + dt] = WMMA_F16(a, bf, acc[m * 4 + dt]);
            }
        __syncthreads();
    }

    const int lane = threadIdx.x & 31;
    const int hi   = lane >> 4;
    const int nn   = lane & 15;
#pragma unroll
    for (int dt = 0; dt < 4; ++dt)
#pragma unroll
        for (int r = 0; r < 8; ++r) {
            const int m = r + 8 * hi;
            const int d = dt * 16 + nn;
            q [((size_t)bh * Tc + t0 + m) * Dh + d] = (half_t)(acc[dt]     [r] * 0.03125f);
            k [((size_t)bh * Tc + t0 + m) * Dh + d] = (half_t)(acc[4 + dt] [r]);
            vt[((size_t)bh * Dh + d) * Tc + t0 + m] = (half_t)(acc[8 + dt] [r]);
        }
}

// ---------------------------------------------------------------------------
// Kernel 2: causal flash attention. grid = (T/64, B*H), block = 128 (4 waves).
// ---------------------------------------------------------------------------
__global__ __launch_bounds__(128) void attn_kernel(
    const half_t* __restrict__ q, const half_t* __restrict__ k,
    const half_t* __restrict__ vt, half_t* __restrict__ obuf) {

    const int wave = threadIdx.x >> 5;
    const int q0   = blockIdx.x * 64 + wave * 16;
    const int bh   = blockIdx.y;
    const int b    = bh >> 4;
    const int h    = bh & 15;

    const half_t* qbase = q  + ((size_t)bh * Tc + q0) * Dh;
    const half_t* kbase = k  + (size_t)bh * Tc * Dh;
    const half_t* vbase = vt + (size_t)bh * Dh * Tc;

    const int lane = threadIdx.x & 31;
    const int hi   = lane >> 4;
    const int nn   = lane & 15;

    const v16h A0 = load_a_f16(qbase,      Dh);
    const v16h A1 = load_a_f16(qbase + 32, Dh);

    v8f O[4];
#pragma unroll
    for (int i = 0; i < 4; ++i) O[i] = zero8();
    float mrow[8], lrow[8];
#pragma unroll
    for (int r = 0; r < 8; ++r) { mrow[r] = -1e30f; lrow[r] = 0.0f; }

    __shared__ half_t Plds[4 * 16 * 32];
    half_t* Pw = Plds + wave * (16 * 32);

    const int kend = q0 + 16;
    for (int j = 0; j < kend; j += 32) {
        v8f S0 = zero8(), S1 = zero8();
        {
            const v16h B0a = load_b_f16(kbase + (size_t)j * Dh,             Dh);
            const v16h B0b = load_b_f16(kbase + (size_t)j * Dh + 32,        Dh);
            S0 = WMMA_F16(A0, B0a, S0);
            S0 = WMMA_F16(A1, B0b, S0);
            const v16h B1a = load_b_f16(kbase + (size_t)(j + 16) * Dh,      Dh);
            const v16h B1b = load_b_f16(kbase + (size_t)(j + 16) * Dh + 32, Dh);
            S1 = WMMA_F16(A0, B1a, S1);
            S1 = WMMA_F16(A1, B1b, S1);
        }

#pragma unroll
        for (int r = 0; r < 8; ++r) {
            const int m  = r + 8 * hi;
            const int qm = q0 + m;
            float s0 = (j + nn      <= qm) ? S0[r] : -1e30f;
            float s1 = (j + 16 + nn <= qm) ? S1[r] : -1e30f;

            float mx = fmaxf(s0, s1);
            mx = fmaxf(mx, __shfl_xor(mx, 1));
            mx = fmaxf(mx, __shfl_xor(mx, 2));
            mx = fmaxf(mx, __shfl_xor(mx, 4));
            mx = fmaxf(mx, __shfl_xor(mx, 8));

            const float mnew  = fmaxf(mrow[r], mx);
            const float alpha = __expf(mrow[r] - mnew);
            mrow[r] = mnew;

            const float p0 = __expf(s0 - mnew);
            const float p1 = __expf(s1 - mnew);
            float rs = p0 + p1;
            rs += __shfl_xor(rs, 1);
            rs += __shfl_xor(rs, 2);
            rs += __shfl_xor(rs, 4);
            rs += __shfl_xor(rs, 8);
            lrow[r] = lrow[r] * alpha + rs;

#pragma unroll
            for (int dt = 0; dt < 4; ++dt) O[dt][r] *= alpha;

            Pw[m * 32 + nn]      = (half_t)p0;
            Pw[m * 32 + 16 + nn] = (half_t)p1;
        }
        lds_fence_wave();

        const v16h Pa = load_a_f16(Pw, 32);
#pragma unroll
        for (int dt = 0; dt < 4; ++dt) {
            const v16h Vb = load_b_f16(vbase + (size_t)(dt * 16) * Tc + j, Tc);
            O[dt] = WMMA_F16(Pa, Vb, O[dt]);
        }
        lds_fence_wave();
    }

#pragma unroll
    for (int r = 0; r < 8; ++r) {
        const float inv = 1.0f / lrow[r];
        const int   m   = r + 8 * hi;
#pragma unroll
        for (int dt = 0; dt < 4; ++dt)
            obuf[((size_t)(b * Tc + q0 + m)) * Cc + h * Dh + dt * 16 + nn] =
                (half_t)(O[dt][r] * inv);
    }
}

// ---------------------------------------------------------------------------
// Kernel 3: out = O @ Wo^T + bo. grid = (B*T/128, C/64), block = 256 (8 waves).
// Woh f16 [n][k]; 64x32 tiles async-copied to LDS, double-buffered
// (1 async b128 per wave per step).
// ---------------------------------------------------------------------------
__global__ __launch_bounds__(256) void proj_kernel(
    const half_t* __restrict__ obuf, const half_t* __restrict__ woh,
    const float* __restrict__ bo, float* __restrict__ out) {

    const int wave = threadIdx.x >> 5;
    const int m0   = blockIdx.x * 128 + wave * 16;
    const int n0   = blockIdx.y * 64;

    constexpr int TILE = 64 * 32;
    __shared__ half_t Wlds[2 * TILE];            // 8KB double buffer

    v8f acc[4];
#pragma unroll
    for (int i = 0; i < 4; ++i) acc[i] = zero8();

    const unsigned lds0 = lds_offset(Wlds);

    auto stage = [&](int buf, int k0) {
        const int f   = threadIdx.x;             // 0..255 chunk id
        const int row = f >> 2;                  // 0..63
        const int seg = f & 3;
        const half_t* g = woh + (size_t)(n0 + row) * Cc + k0 + seg * 8;
        async_copy_b128(lds0 + (unsigned)(buf * TILE + row * 32 + seg * 8) * 2, g);
    };

    stage(0, 0);
    for (int s = 0; s < 32; ++s) {
        if (s + 1 < 32) { stage((s + 1) & 1, (s + 1) * 32); wait_async_le1(); }
        else            { wait_async_0(); }
        __syncthreads();

        const half_t* Wcur = Wlds + (s & 1) * TILE;
        const v16h a = load_a_f16(obuf + (size_t)m0 * Cc + s * 32, Cc);
#pragma unroll
        for (int dt = 0; dt < 4; ++dt) {
            const v16h bf = load_b_f16(Wcur + dt * 16 * 32, 32);
            acc[dt] = WMMA_F16(a, bf, acc[dt]);
        }
        __syncthreads();
    }

    const int lane = threadIdx.x & 31;
    const int hi   = lane >> 4;
    const int nn   = lane & 15;
#pragma unroll
    for (int dt = 0; dt < 4; ++dt)
#pragma unroll
        for (int r = 0; r < 8; ++r) {
            const int m = m0 + r + 8 * hi;
            const int n = n0 + dt * 16 + nn;
            out[(size_t)m * Cc + n] = acc[dt][r] + bo[n];
        }
}

// ---------------------------------------------------------------------------
extern "C" void kernel_launch(void* const* d_in, const int* in_sizes, int n_in,
                              void* d_out, int out_size, void* d_ws, size_t ws_size,
                              hipStream_t stream) {
    const float* x  = (const float*)d_in[0];
    const float* Wq = (const float*)d_in[1];
    const float* Wk = (const float*)d_in[2];
    const float* Wv = (const float*)d_in[3];
    const float* Wo = (const float*)d_in[4];
    const float* bo = (const float*)d_in[5];
    float* out = (float*)d_out;

    const size_t qkv_elems = (size_t)Bc * Hc * Tc * Dh;   // 4M halves
    const size_t x_elems   = (size_t)Bc * Tc * Cc;        // 4M halves
    const size_t w_elems   = (size_t)Hc * Cc * Dh;        // 1M halves per mat

    half_t* q    = (half_t*)d_ws;
    half_t* k    = q    + qkv_elems;
    half_t* vt   = k    + qkv_elems;
    half_t* obuf = vt   + qkv_elems;                      // [B*T][C]
    half_t* xh   = obuf + x_elems;                        // [B*T][C]
    half_t* wall = xh   + x_elems;                        // [3][H][64][1024]
    half_t* woh  = wall + 3 * w_elems;                    // [1024][1024]

    // one-shot conversions / transposes (weights+x ~ 34MB read: trivial)
    cvt_f16_kernel<<<(int)(x_elems / 256), 256, 0, stream>>>(x, xh, (int)x_elems);
    transpose_w_kernel<<<(int)(w_elems / 256), 256, 0, stream>>>(Wq, wall);
    transpose_w_kernel<<<(int)(w_elems / 256), 256, 0, stream>>>(Wk, wall + w_elems);
    transpose_w_kernel<<<(int)(w_elems / 256), 256, 0, stream>>>(Wv, wall + 2 * w_elems);
    cvt_f16_kernel<<<(int)(Cc * Cc / 256), 256, 0, stream>>>(Wo, woh, Cc * Cc);

    qkv_kernel <<<dim3(Bc * Tc / 128, Hc), 256, 0, stream>>>(xh, wall, q, k, vt);
    attn_kernel<<<dim3(Tc / 64, Bc * Hc), 128, 0, stream>>>(q, k, vt, obuf);
    proj_kernel<<<dim3(Bc * Tc / 128, Cc / 64), 256, 0, stream>>>(obuf, woh, bo, out);
}
